// PSN_14568529068276
// MI455X (gfx1250) — matrix operand order repaired and checked by reading, not compiled, not run on hardware
//
#include <hip/hip_runtime.h>

// ---------------------------------------------------------------------------
// Fused VQ quantizer for MI455X (gfx1250, wave32, WMMA).
//   dist GEMM (16384x8192x64) fused with row-argmin using 3xBF16 ~= FP32
//   decomposition on V_WMMA_F32_16X16X32_BF16. Codebook tiles are staged in
//   LDS (double buffered) via GLOBAL_LOAD_ASYNC_TO_LDS_B128 when available.
// ---------------------------------------------------------------------------

typedef __bf16 bf16;
typedef __attribute__((ext_vector_type(16))) __bf16 v16bf;
typedef __attribute__((ext_vector_type(8)))  float  v8f;
typedef int v4i_gcc __attribute__((vector_size(16)));   // builtin's pointee type
typedef __attribute__((address_space(1))) v4i_gcc g_v4i;
typedef __attribute__((address_space(3))) v4i_gcc l_v4i;

#define N_ROWS  16384   // B*E = 32*512
#define N_CODES 8192    // O
#define CH      64      // C
#define QNOISE  0.5f
#define BETA    0.25f

#define TILE_CODES 32                       // codes staged per step
#define NSTEP      (N_CODES / TILE_CODES)   // 256

#if defined(__has_builtin)
#if __has_builtin(__builtin_amdgcn_global_load_async_to_lds_b128) && \
    __has_builtin(__builtin_amdgcn_s_wait_asynccnt)
#define VQ_USE_ASYNC 1
#endif
#endif

// --------------------------------------------------------------------------
// Kernel 1: codebook prep — split f32 codebook into bf16 hi/lo planes and
// compute per-code squared norms (xnorm is row-constant -> dropped).
// --------------------------------------------------------------------------
__global__ void __launch_bounds__(256) vq_prep(const float* __restrict__ cb,
                                               bf16* __restrict__ cb_hi,
                                               bf16* __restrict__ cb_lo,
                                               float* __restrict__ cnorm) {
    int k = blockIdx.x * blockDim.x + threadIdx.x;
    if (k >= N_CODES) return;
    const float* row = cb + (size_t)k * CH;
    float s = 0.0f;
    for (int c = 0; c < CH; ++c) {
        float x = row[c];
        s += x * x;
        bf16 hi = (bf16)x;
        float lo = x - (float)hi;
        cb_hi[(size_t)k * CH + c] = hi;
        cb_lo[(size_t)k * CH + c] = (bf16)lo;
    }
    cnorm[k] = s;
}

// --------------------------------------------------------------------------
// Kernel 2: fused GEMM + argmin.
//   Block = 256 threads (8 waves) = 128 rows; wave owns a 16-row A tile held
//   in registers (bf16 hi/lo WMMA layout). All 8 waves share LDS-staged
//   codebook tiles of 32 codes (hi+lo planes), double buffered.
// A layout (16-bit A 16x32, contiguous k, not ISA table order): lane (h=lane>>4, m=lane&15):
//   elem e -> channel 32kb + 16h + e (contiguous k order)
// B layout (32x16): lane (h, n): elem e -> channel kb + 16h + e of code n
// D layout: lane (h, n), VGPR v -> D[m = v + 8h][n]
// --------------------------------------------------------------------------
__global__ void __launch_bounds__(256) vq_argmin(const float* __restrict__ flat,
                                                 const bf16* __restrict__ cb_hi,
                                                 const bf16* __restrict__ cb_lo,
                                                 const float* __restrict__ cnorm,
                                                 int* __restrict__ inds) {
    const int tid     = threadIdx.x;
    const int lane    = tid & 31;
    const int wave    = tid >> 5;
    const int h       = lane >> 4;
    const int n       = lane & 15;
    const int rowBase = (blockIdx.x * 8 + wave) * 16;

    // [buf][plane(hi/lo)][code][ch]  -> 2*2*32*64 bf16 = 16 KiB
    __shared__ __align__(32) bf16 sB[2][2][TILE_CODES * CH];

    // ---- Load A tile (16 rows x 64 ch) into bf16 hi/lo WMMA register layout
    v16bf a_hi[2], a_lo[2];
    {
        const float* arow = flat + (size_t)(rowBase + n) * CH;
        for (int kb = 0; kb < 2; ++kb) {
            const int base0 = kb * 32 + 16 * h;       // e = 0..7
            const int base1 = kb * 32 + 16 * h + 8;   // e = 8..15
            for (int e = 0; e < 8; ++e) {
                float x0 = arow[base0 + e];
                float x1 = arow[base1 + e];
                bf16 h0 = (bf16)x0, h1 = (bf16)x1;
                a_hi[kb][e]     = h0;
                a_lo[kb][e]     = (bf16)(x0 - (float)h0);
                a_hi[kb][e + 8] = h1;
                a_lo[kb][e + 8] = (bf16)(x1 - (float)h1);
            }
        }
    }

    float minv[8];
    int   mini[8];
#pragma unroll
    for (int v = 0; v < 8; ++v) { minv[v] = 3.4e38f; mini[v] = 0; }

    // Cooperative tile stage: 4 KiB/plane = 256 threads x 16B each.
    // Thread tid copies bf16 elements [tid*8, tid*8+8) of the tile.
    auto stage = [&](int step, int buf) {
        const size_t srcOff = (size_t)step * TILE_CODES * CH + (size_t)tid * 8;
#ifdef VQ_USE_ASYNC
        {
            void* sh = (void*)(uintptr_t)(cb_hi + srcOff);
            void* sl = (void*)(uintptr_t)(cb_lo + srcOff);
            __builtin_amdgcn_global_load_async_to_lds_b128(
                (g_v4i*)sh, (l_v4i*)(void*)&sB[buf][0][tid * 8], 0, 0);
            __builtin_amdgcn_global_load_async_to_lds_b128(
                (g_v4i*)sl, (l_v4i*)(void*)&sB[buf][1][tid * 8], 0, 0);
        }
#else
        *(uint4*)&sB[buf][0][tid * 8] = *(const uint4*)(cb_hi + srcOff);
        *(uint4*)&sB[buf][1][tid * 8] = *(const uint4*)(cb_lo + srcOff);
#endif
    };
    auto wait_stage = [&]() {
#ifdef VQ_USE_ASYNC
        __builtin_amdgcn_s_wait_asynccnt(0);
#endif
    };

    stage(0, 0);
    for (int step = 0; step < NSTEP; ++step) {
        const int cur = step & 1;
        wait_stage();          // this wave's async tile loads are done
        __syncthreads();       // all waves' loads visible; prev compute done
        if (step + 1 < NSTEP) stage(step + 1, cur ^ 1);

        const int codeBase = step * TILE_CODES;
#pragma unroll
        for (int sub = 0; sub < 2; ++sub) {
            const int codeLocal = sub * 16 + n;
            const int code      = codeBase + codeLocal;
            const bf16* bh = &sB[cur][0][codeLocal * CH + 16 * h];
            const bf16* bl = &sB[cur][1][codeLocal * CH + 16 * h];

            v8f acc = {};
#pragma unroll
            for (int kb = 0; kb < 2; ++kb) {
                v16bf bhi = *(const v16bf*)(bh + kb * 32);
                v16bf blo = *(const v16bf*)(bl + kb * 32);
                acc = __builtin_amdgcn_wmma_f32_16x16x32_bf16(
                    false, a_hi[kb], false, bhi, (short)0, acc, false, false);
                acc = __builtin_amdgcn_wmma_f32_16x16x32_bf16(
                    false, a_hi[kb], false, blo, (short)0, acc, false, false);
                acc = __builtin_amdgcn_wmma_f32_16x16x32_bf16(
                    false, a_lo[kb], false, bhi, (short)0, acc, false, false);
            }

            const float cn = cnorm[code];
#pragma unroll
            for (int v = 0; v < 8; ++v) {
                float d = cn - 2.0f * acc[v];
                if (d < minv[v]) { minv[v] = d; mini[v] = code; }
            }
        }
    }

    // ---- reduce (min, idx) across the 16 lanes of each half (same rows)
#pragma unroll
    for (int off = 8; off >= 1; off >>= 1) {
#pragma unroll
        for (int v = 0; v < 8; ++v) {
            float ov = __shfl_xor(minv[v], off, 16);
            int   oi = __shfl_xor(mini[v], off, 16);
            if (ov < minv[v] || (ov == minv[v] && oi < mini[v])) {
                minv[v] = ov;
                mini[v] = oi;
            }
        }
    }
    if (n == 0) {
#pragma unroll
        for (int v = 0; v < 8; ++v) inds[rowBase + 8 * h + v] = mini[v];
    }
}

// --------------------------------------------------------------------------
// Kernel 3: gather q_det / q_noisy + MSE partial sums. Wave handles 8 rows,
// lanes stride channels (c = lane, lane+32) -> coalesced 128B row accesses.
// out = latents + sg(q_noisy - latents) == q_noisy numerically.
// --------------------------------------------------------------------------
__global__ void __launch_bounds__(256) vq_gather_loss(const float* __restrict__ flat,
                                                      const float* __restrict__ y,
                                                      const float* __restrict__ cb,
                                                      const float* __restrict__ noise,
                                                      const int* __restrict__ inds,
                                                      float* __restrict__ out,
                                                      float* __restrict__ partials) {
    const int wave    = threadIdx.x >> 5;
    const int lane    = threadIdx.x & 31;
    const int rowBase = (blockIdx.x * 8 + wave) * 8;   // 8 rows per wave

    float recon = 0.0f, commit = 0.0f, cbl = 0.0f;
#pragma unroll
    for (int i = 0; i < 8; ++i) {
        const int r  = rowBase + i;
        const int id = inds[r];
        int in = id + __float2int_rn(noise[r] * QNOISE);   // round-half-even
        in = in < 0 ? 0 : (in > N_CODES - 1 ? N_CODES - 1 : in);

        const float* qd = cb + (size_t)id * CH;
        const float* qn = cb + (size_t)in * CH;
        const float* xr = flat + (size_t)r * CH;
        const float* yr = y + (size_t)r * CH;
        float* orow = out + (size_t)r * CH;

#pragma unroll
        for (int cc = 0; cc < 2; ++cc) {
            const int c = lane + cc * 32;
            float qnv = qn[c], qdv = qd[c], xv = xr[c], yv = yr[c];
            orow[c] = qnv;
            float dr = qnv - yv;  recon  += dr * dr;
            float dc = xv - qdv;  commit += dc * dc;
            float dq = qnv - xv;  cbl    += dq * dq;
        }
    }

    // wave reduce then cross-wave via LDS
    float vals[3] = {recon, commit, cbl};
#pragma unroll
    for (int i = 0; i < 3; ++i) {
        float v = vals[i];
        for (int off = 16; off >= 1; off >>= 1) v += __shfl_xor(v, off, 32);
        vals[i] = v;
    }
    __shared__ float smem[8][3];
    if (lane == 0) {
        smem[wave][0] = vals[0]; smem[wave][1] = vals[1]; smem[wave][2] = vals[2];
    }
    __syncthreads();
    if (threadIdx.x == 0) {
        float s0 = 0.f, s1 = 0.f, s2 = 0.f;
        for (int w = 0; w < 8; ++w) { s0 += smem[w][0]; s1 += smem[w][1]; s2 += smem[w][2]; }
        partials[blockIdx.x * 3 + 0] = s0;
        partials[blockIdx.x * 3 + 1] = s1;
        partials[blockIdx.x * 3 + 2] = s2;
    }
}

// --------------------------------------------------------------------------
// Kernel 4: combine block partials into the scalar loss (fixed order).
// --------------------------------------------------------------------------
__global__ void vq_finalize(const float* __restrict__ partials,
                            float* __restrict__ loss_out, int nblocks) {
    if (blockIdx.x == 0 && threadIdx.x == 0) {
        float r = 0.f, c = 0.f, q = 0.f;
        for (int b = 0; b < nblocks; ++b) {
            r += partials[3 * b + 0];
            c += partials[3 * b + 1];
            q += partials[3 * b + 2];
        }
        const float inv = 1.0f / (float)((size_t)N_ROWS * CH);
        *loss_out = (r + q + BETA * c) * inv;
    }
}

// --------------------------------------------------------------------------
extern "C" void kernel_launch(void* const* d_in, const int* in_sizes, int n_in,
                              void* d_out, int out_size, void* d_ws, size_t ws_size,
                              hipStream_t stream) {
    const float* latents = (const float*)d_in[0];   // [32,512,64] == flat [16384,64]
    const float* y       = (const float*)d_in[1];
    const float* cb      = (const float*)d_in[2];   // [8192,64]
    const float* noise   = (const float*)d_in[3];   // [16384]

    float* out  = (float*)d_out;          // 16384*64 floats
    float* loss = out + (size_t)N_ROWS * CH;

    char* ws = (char*)d_ws;
    bf16*  cb_hi    = (bf16*)(ws);                                  // 1 MiB
    bf16*  cb_lo    = (bf16*)(ws + (1u << 20));                     // 1 MiB
    float* cnorm    = (float*)(ws + (2u << 20));                    // 32 KiB
    int*   inds     = (int*)(ws + (2u << 20) + (32u << 10));        // 64 KiB
    float* partials = (float*)(ws + (2u << 20) + (96u << 10));      // 256*3 f32

    vq_prep<<<N_CODES / 256, 256, 0, stream>>>(cb, cb_hi, cb_lo, cnorm);
    vq_argmin<<<N_ROWS / 128, 256, 0, stream>>>(latents, cb_hi, cb_lo, cnorm, inds);
    const int gblocks = N_ROWS / 64;    // wave=8 rows, 8 waves/block
    vq_gather_loss<<<gblocks, 256, 0, stream>>>(latents, y, cb, noise, inds, out, partials);
    vq_finalize<<<1, 32, 0, stream>>>(partials, loss, gblocks);
}
